// Model_15436112462035
// MI455X (gfx1250) — matrix operand order repaired and checked by reading, not compiled, not run on hardware
//
#include <hip/hip_runtime.h>
#include <hip/hip_bf16.h>
#include <math.h>

// Problem constants (fixed by the reference)
#define BB 128
#define UU 16
#define DD 256
#define NN 1024
#define OO 256

typedef __attribute__((ext_vector_type(8)))  float  v8f;
typedef __attribute__((ext_vector_type(8)))  __bf16 v8bf;
typedef __attribute__((ext_vector_type(16))) __bf16 v16bf;

#define AST 40   // LDS row stride in bf16 (80 B: 16B-aligned, 20-dword bank skew)

static __device__ __forceinline__ v16bf cat16(v8bf lo, v8bf hi) {
    return __builtin_shufflevector(lo, hi, 0,1,2,3,4,5,6,7,8,9,10,11,12,13,14,15);
}
static __device__ __forceinline__ unsigned pack_bf16(float lo, float hi) {
    unsigned short l = __builtin_bit_cast(unsigned short, (__bf16)lo);
    unsigned short h = __builtin_bit_cast(unsigned short, (__bf16)hi);
    return ((unsigned)h << 16) | l;
}

// ---------------------------------------------------------------------------
// Kernel 1: lr[b,u] = softmax_u( dot(X[b,u,:], alr[u,:]) / T )
// ---------------------------------------------------------------------------
__global__ __launch_bounds__(512) void lr_kernel(const float* __restrict__ X,
                                                 const float* __restrict__ alr,
                                                 const float* __restrict__ temp,
                                                 float* __restrict__ lr_out) {
    __shared__ float logits[UU];
    const int b    = blockIdx.x;
    const int u    = threadIdx.x >> 5;   // one wave per u
    const int lane = threadIdx.x & 31;

    const float* xp = X   + ((size_t)b * UU + u) * DD;
    const float* ap = alr + (size_t)u * DD;
    float s = 0.f;
#pragma unroll
    for (int i = lane; i < DD; i += 32) s += xp[i] * ap[i];
#pragma unroll
    for (int off = 16; off > 0; off >>= 1) s += __shfl_xor(s, off, 32);
    if (lane == 0) logits[u] = s / temp[0];
    __syncthreads();

    if (threadIdx.x == 0) {
        float mx = -3.4e38f;
#pragma unroll
        for (int k = 0; k < UU; ++k) mx = fmaxf(mx, logits[k]);
        float e[UU]; float sum = 0.f;
#pragma unroll
        for (int k = 0; k < UU; ++k) { e[k] = __expf(logits[k] - mx); sum += e[k]; }
        const float inv = 1.f / sum;
#pragma unroll
        for (int k = 0; k < UU; ++k) lr_out[b * UU + k] = e[k] * inv;
    }
}

// ---------------------------------------------------------------------------
// Kernel 2: new_state = (1-lr)*s + lr*tanh( state@(W[u]*sr) + X@Win[u] + bias )
// Block tile 128x128, 8 waves in a 4(M) x 2(N) grid; wave tile 32x64 -> 8 WMMA
// per K-step. Double-buffered LDS; f32->bf16 conversion while staging.
// K loop: 32 steps over W (A=state, scaled by sr), then 8 steps over Win (A=X).
// ---------------------------------------------------------------------------
__global__ __launch_bounds__(256) void state_kernel(const float* __restrict__ state,
                                                    const float* __restrict__ X,
                                                    const float* __restrict__ W,
                                                    const float* __restrict__ Win,
                                                    const float* __restrict__ bias,
                                                    const float* __restrict__ sr,
                                                    const float* __restrict__ lrw,
                                                    float* __restrict__ ns_out,
                                                    __bf16* __restrict__ ns_bf) {
    __shared__ __bf16 As[2][BB * AST];    // A tile: 128 rows x 32 K
    __shared__ __bf16 Bs[2][128 * AST];   // B tile transposed: 128 cols x 32 K

    const int u    = blockIdx.y;
    const int nb   = blockIdx.x * 128;           // N-tile base
    const int t    = threadIdx.x;
    const int wave = t >> 5;
    const int lane = t & 31;
    const float srv = sr[u];

    // loader coordinates
    const int rA = t >> 1,        cA = (t & 1) * 16;  // A: 16 f32, one row chunk
    const int kB = (t >> 4) << 1, cB = t & 15;        // B: 2 K-rows x 8 strided cols

    unsigned apk[8], bpk[8];

    auto stage = [&](int kk) {
        const float* aptr; const float* bsrc; float scale;
        if (kk < 32) {
            const int k0 = kk * 32;
            aptr  = state + ((size_t)rA * UU + u) * NN + k0 + cA;
            bsrc  = W + (size_t)u * NN * NN + (size_t)(k0 + kB) * NN + nb + cB;
            scale = srv;
            if (kk < 31) __builtin_prefetch(bsrc + 32 * NN, 0, 1);
        } else {
            const int k0 = (kk - 32) * 32;
            aptr  = X + ((size_t)rA * UU + u) * DD + k0 + cA;
            bsrc  = Win + (size_t)u * DD * NN + (size_t)(k0 + kB) * NN + nb + cB;
            scale = 1.f;
        }
        float4 a0 = ((const float4*)aptr)[0];
        float4 a1 = ((const float4*)aptr)[1];
        float4 a2 = ((const float4*)aptr)[2];
        float4 a3 = ((const float4*)aptr)[3];
        apk[0] = pack_bf16(a0.x, a0.y); apk[1] = pack_bf16(a0.z, a0.w);
        apk[2] = pack_bf16(a1.x, a1.y); apk[3] = pack_bf16(a1.z, a1.w);
        apk[4] = pack_bf16(a2.x, a2.y); apk[5] = pack_bf16(a2.z, a2.w);
        apk[6] = pack_bf16(a3.x, a3.y); apk[7] = pack_bf16(a3.z, a3.w);
#pragma unroll
        for (int i = 0; i < 8; ++i) {
            float f0 = bsrc[16 * i];
            float f1 = bsrc[16 * i + NN];
            bpk[i] = pack_bf16(f0 * scale, f1 * scale);
        }
    };

    auto commit = [&](int buf) {
        uint4* ad = (uint4*)&As[buf][rA * AST + cA];
        ad[0] = make_uint4(apk[0], apk[1], apk[2], apk[3]);
        ad[1] = make_uint4(apk[4], apk[5], apk[6], apk[7]);
#pragma unroll
        for (int i = 0; i < 8; ++i)
            *(unsigned*)&Bs[buf][(cB + 16 * i) * AST + kB] = bpk[i];  // bank-conflict-free b32
    };

    v8f acc[2][4] = {};
    const int m0 = (wave >> 1) * 32;       // wave row base   (4 waves in M)
    const int nw = (wave & 1) * 64;        // wave col base   (2 waves in N)
    const int rf = lane & 15;
    const int ka = (lane & 16) ? 8 : 0;    // A frag K interleave (ISA layout)
    const int kb = (lane & 16) ? 8 : 0;    // B frag K interleave (ISA layout)

    stage(0); commit(0);

    for (int kk = 0; kk < 40; ++kk) {
        __syncthreads();
        if (kk < 39) stage(kk + 1);

        const int buf = kk & 1;
        v16bf afr[2];
#pragma unroll
        for (int i = 0; i < 2; ++i) {
            const int row = m0 + 16 * i + rf;
            v8bf lo = *(const v8bf*)&As[buf][row * AST + ka];
            v8bf hi = *(const v8bf*)&As[buf][row * AST + ka + 16];
            afr[i] = cat16(lo, hi);
        }
#pragma unroll
        for (int j = 0; j < 4; ++j) {
            const int col = nw + 16 * j + rf;
            v8bf lo = *(const v8bf*)&Bs[buf][col * AST + kb];
            v8bf hi = *(const v8bf*)&Bs[buf][col * AST + kb + 16];
            v16bf bfr = cat16(lo, hi);
#pragma unroll
            for (int i = 0; i < 2; ++i)
                acc[i][j] = __builtin_amdgcn_wmma_f32_16x16x32_bf16(
                    false, afr[i], false, bfr, (short)0, acc[i][j], false, false);
        }
        if (kk < 39) commit((kk + 1) & 1);
    }

    // epilogue: bias + tanh + lr blend; write f32 new_state and bf16 copy
    const int m_add = (lane & 16) ? 8 : 0;
#pragma unroll
    for (int i = 0; i < 2; ++i) {
#pragma unroll
        for (int j = 0; j < 4; ++j) {
#pragma unroll
            for (int r = 0; r < 8; ++r) {
                const int b = m0 + 16 * i + r + m_add;     // batch index
                const int n = nb + nw + 16 * j + rf;       // state column
                const size_t idx = ((size_t)b * UU + u) * NN + n;
                const float lrv = lrw[b * UU + u];
                const float sv  = state[idx];
                const float tv  = tanhf(acc[i][j][r] + bias[(size_t)u * NN + n]);
                const float nsv = (1.f - lrv) * sv + lrv * tv;
                ns_out[idx] = nsv;
                ns_bf[idx]  = (__bf16)nsv;
            }
        }
    }
}

// ---------------------------------------------------------------------------
// Kernel 3: output = new_state @ Wout[u]  (A already bf16 in workspace)
// Block tile 128x64, 8 waves each 16x64; double-buffered.
// ---------------------------------------------------------------------------
__global__ __launch_bounds__(256) void out_kernel(const __bf16* __restrict__ nsb,
                                                  const float* __restrict__ Wout,
                                                  float* __restrict__ out) {
    __shared__ __bf16 As[2][BB * AST];
    __shared__ __bf16 Bs[2][64 * AST];

    const int u    = blockIdx.y;
    const int ob   = blockIdx.x * 64;            // O-tile base
    const int t    = threadIdx.x;
    const int wave = t >> 5;
    const int lane = t & 31;

    const int rA = t >> 1,        cA = (t & 1) * 16;
    const int kB = (t >> 4) << 1, cB = t & 15;

    uint4 av0, av1; unsigned bpk[4];

    auto stage = [&](int kk) {
        const int k0 = kk * 32;
        const __bf16* aptr = nsb + ((size_t)rA * UU + u) * NN + k0 + cA;
        av0 = ((const uint4*)aptr)[0];
        av1 = ((const uint4*)aptr)[1];
        const float* bsrc = Wout + (size_t)u * NN * OO + (size_t)(k0 + kB) * OO + ob + cB;
        if (kk < 31) __builtin_prefetch(bsrc + 32 * OO, 0, 1);
#pragma unroll
        for (int i = 0; i < 4; ++i) {
            float f0 = bsrc[16 * i];
            float f1 = bsrc[16 * i + OO];
            bpk[i] = pack_bf16(f0, f1);
        }
    };
    auto commit = [&](int buf) {
        *(uint4*)&As[buf][rA * AST + cA]     = av0;
        *(uint4*)&As[buf][rA * AST + cA + 8] = av1;
#pragma unroll
        for (int i = 0; i < 4; ++i)
            *(unsigned*)&Bs[buf][(cB + 16 * i) * AST + kB] = bpk[i];
    };

    v8f acc[4] = {};
    const int m0 = 16 * wave;
    const int rf = lane & 15;
    const int ka = (lane & 16) ? 8 : 0;
    const int kb = (lane & 16) ? 8 : 0;

    stage(0); commit(0);

    for (int kk = 0; kk < 32; ++kk) {
        __syncthreads();
        if (kk < 31) stage(kk + 1);

        const int buf = kk & 1;
        const int row = m0 + rf;
        v8bf alo = *(const v8bf*)&As[buf][row * AST + ka];
        v8bf ahi = *(const v8bf*)&As[buf][row * AST + ka + 16];
        v16bf afr = cat16(alo, ahi);
#pragma unroll
        for (int j = 0; j < 4; ++j) {
            const int col = 16 * j + rf;
            v8bf blo = *(const v8bf*)&Bs[buf][col * AST + kb];
            v8bf bhi = *(const v8bf*)&Bs[buf][col * AST + kb + 16];
            v16bf bfr = cat16(blo, bhi);
            acc[j] = __builtin_amdgcn_wmma_f32_16x16x32_bf16(
                false, afr, false, bfr, (short)0, acc[j], false, false);
        }
        if (kk < 31) commit((kk + 1) & 1);
    }

    const int m_add = (lane & 16) ? 8 : 0;
#pragma unroll
    for (int j = 0; j < 4; ++j) {
#pragma unroll
        for (int r = 0; r < 8; ++r) {
            const int b = m0 + r + m_add;
            const int o = ob + 16 * j + rf;
            out[((size_t)b * UU + u) * OO + o] = acc[j][r];
        }
    }
}

// ---------------------------------------------------------------------------
extern "C" void kernel_launch(void* const* d_in, const int* in_sizes, int n_in,
                              void* d_out, int out_size, void* d_ws, size_t ws_size,
                              hipStream_t stream) {
    const float* X     = (const float*)d_in[0];
    const float* state = (const float*)d_in[1];
    const float* W     = (const float*)d_in[2];
    const float* Win   = (const float*)d_in[3];
    const float* bias  = (const float*)d_in[4];
    const float* Wout  = (const float*)d_in[5];
    const float* sr    = (const float*)d_in[6];
    const float* alr   = (const float*)d_in[7];
    const float* temp  = (const float*)d_in[8];

    float*  lr  = (float*)d_ws;                                  // B*U f32 (8 KB)
    __bf16* nsb = (__bf16*)((char*)d_ws + 8192);                 // B*U*N bf16 (4 MB)

    float* ns_out  = (float*)d_out;                              // new_state (B,U,N)
    float* out_out = (float*)d_out + (size_t)BB * UU * NN;       // output   (B,U,O)

    lr_kernel<<<BB, 512, 0, stream>>>(X, alr, temp, lr);
    state_kernel<<<dim3(NN / 128, UU), 256, 0, stream>>>(
        state, X, W, Win, bias, sr, lr, ns_out, nsb);
    out_kernel<<<dim3(OO / 64, UU), 256, 0, stream>>>(nsb, Wout, out_out);
}
